// SFFAM_34248069218976
// MI455X (gfx1250) — matrix operand order repaired and checked by reading, not compiled, not run on hardware
//
#include <hip/hip_runtime.h>

// ---------------------------------------------------------------------------
// Problem constants (from reference)
// ---------------------------------------------------------------------------
#define HH_   1408
#define WW_   1408
#define HWl_  (HH_*WW_)            // 1,982,464 ; max key = 4*HW-1 < 2^23
#define N1_   250000
#define N2_   300000
#define C_    256
#define GROUPS_ ((N1_ + 31) / 32)  // 7813 groups of 32 rows (last has 16)
#define WPB_  2                    // waves per block

typedef __bf16 v16bf __attribute__((ext_vector_type(16)));
typedef __bf16 v8bf  __attribute__((ext_vector_type(8)));
typedef float  v8f   __attribute__((ext_vector_type(8)));
typedef float  v4f   __attribute__((ext_vector_type(4)));

union BF16x16 { v16bf v; v8bf h[2]; __bf16 e[16]; };

// workspace layout (bytes)
#define WS_OFF_TABLE 0u
#define WS_OFF_W1B   (32u*1024u*1024u)                 // table needs 31.72 MB
#define WS_OFF_W2B   (WS_OFF_W1B + 512u*256u*2u)
#define WS_OFF_W3B   (WS_OFF_W2B + 256u*256u*2u)

// ---------------------------------------------------------------------------
// Wave-private LDS fence: DS ops from one wave execute in order; stop the
// compiler from reordering and drain DScnt before cross-lane reads.
// ---------------------------------------------------------------------------
__device__ __forceinline__ void lds_fence() {
    __builtin_amdgcn_wave_barrier();
    asm volatile("s_wait_dscnt 0x0" ::: "memory");
    __builtin_amdgcn_wave_barrier();
}

// ---------------------------------------------------------------------------
// Prep kernel 1: direct-index lookup table  T[key2[j]] = j
// ---------------------------------------------------------------------------
__global__ void build_table_kernel(const int* __restrict__ coords2,
                                   unsigned* __restrict__ T) {
    int j = blockIdx.x * 256 + threadIdx.x;
    if (j < N2_) {
        unsigned b = (unsigned)coords2[3*j + 0];
        unsigned y = (unsigned)coords2[3*j + 1];
        unsigned x = (unsigned)coords2[3*j + 2];
        unsigned key = b * (unsigned)HWl_ + y * (unsigned)WW_ + x;
        T[key] = (unsigned)j;
    }
}

// ---------------------------------------------------------------------------
// Prep kernel 2: f32 -> bf16 weights (row-major k x n: B-fragment layout
// wants lane = N column, 16 K values per lane)
// ---------------------------------------------------------------------------
__global__ void cvt_weights_kernel(const float* __restrict__ W1,
                                   const float* __restrict__ W2,
                                   const float* __restrict__ W3,
                                   __bf16* __restrict__ W1b,
                                   __bf16* __restrict__ W2b,
                                   __bf16* __restrict__ W3b) {
    int i = blockIdx.x * 256 + threadIdx.x;
    if (i < 512*256) W1b[i] = (__bf16)W1[i];
    if (i < 256*256) { W2b[i] = (__bf16)W2[i]; W3b[i] = (__bf16)W3[i]; }
}

// ---------------------------------------------------------------------------
// Fragment loaders
// ---------------------------------------------------------------------------
template <bool NT>
__device__ __forceinline__ v4f ld4(const float* __restrict__ p) {
    if (NT) return __builtin_nontemporal_load((const v4f*)p);
    return *(const v4f*)p;
}

// A (16x32 bf16) from an f32 row: elements 0..7  = K k0..k0+7,
//                                 elements 8..15 = K k0+16..k0+23
template <bool NT>
__device__ __forceinline__ v16bf load_a_f32(const float* __restrict__ row, int k0) {
    v4f a = ld4<NT>(row + k0);
    v4f b = ld4<NT>(row + k0 + 4);
    v4f c = ld4<NT>(row + k0 + 16);
    v4f d = ld4<NT>(row + k0 + 20);
    BF16x16 u;
#pragma unroll
    for (int i = 0; i < 4; ++i) {
        u.e[i]      = (__bf16)a[i];
        u.e[4 + i]  = (__bf16)b[i];
        u.e[8 + i]  = (__bf16)c[i];
        u.e[12 + i] = (__bf16)d[i];
    }
    return u.v;
}

// B (32x16 bf16) from row-major bf16 weights: lane = N column (lane%16),
// elements 0..7 = K 8h..8h+7, elements 8..15 = K 16+8h..16+8h+7 (h = lane/16).
__device__ __forceinline__ v16bf load_b(const __bf16* __restrict__ Wb,
                                        int kbase, int nbase, int lane) {
    const int h = lane >> 4;
    const __bf16* p = Wb + (size_t)kbase * C_ + nbase + (lane & 15);
    BF16x16 u;
#pragma unroll
    for (int s = 0; s < 16; ++s) {
        const int k = (s < 8) ? (8 * h + s) : (16 + 8 * h + (s - 8));
        u.e[s] = p[(size_t)k * C_];
    }
    return u.v;
}

__device__ __forceinline__ v8f wmma_bf16(v16bf a, v16bf b, v8f c) {
    return __builtin_amdgcn_wmma_f32_16x16x32_bf16(false, a, false, b,
                                                   (short)0, c, false, false);
}

// One K=256 GEMM pass from an LDS A-tile (32 rows), B double-buffered,
// each B fragment shared by the two M-tiles.
__device__ __forceinline__ void gemm_lds(const __bf16 (*__restrict__ sf)[264],
                                         const __bf16* __restrict__ Wb,
                                         int lane, int laneRow, int laneHalf,
                                         v8f acc0[16], v8f acc1[16]) {
    for (int s = 0; s < 8; ++s) {
        const int kbase = s * 32;
        BF16x16 u0, u1;
        u0.h[0] = *(const v8bf*)&sf[laneRow][kbase + laneHalf * 8];
        u0.h[1] = *(const v8bf*)&sf[laneRow][kbase + 16 + laneHalf * 8];
        u1.h[0] = *(const v8bf*)&sf[16 + laneRow][kbase + laneHalf * 8];
        u1.h[1] = *(const v8bf*)&sf[16 + laneRow][kbase + 16 + laneHalf * 8];
        v16bf bcur = load_b(Wb, kbase, 0, lane);
#pragma unroll
        for (int nt = 0; nt < 16; ++nt) {
            v16bf bnext = bcur;
            if (nt < 15) bnext = load_b(Wb, kbase, (nt + 1) * 16, lane);
            acc0[nt] = wmma_bf16(u0.v, bcur, acc0[nt]);
            acc1[nt] = wmma_bf16(u1.v, bcur, acc1[nt]);
            bcur = bnext;
        }
    }
}

__device__ __forceinline__ void relu_store_lds(__bf16 (*__restrict__ sf)[264],
                                               const v8f acc0[16], const v8f acc1[16],
                                               int laneRow, int laneHalf) {
#pragma unroll
    for (int nt = 0; nt < 16; ++nt)
#pragma unroll
        for (int r = 0; r < 8; ++r) {
            float v0 = acc0[nt][r]; v0 = v0 > 0.f ? v0 : 0.f;
            float v1 = acc1[nt][r]; v1 = v1 > 0.f ? v1 : 0.f;
            sf[r + 8 * laneHalf][nt * 16 + laneRow]      = (__bf16)v0;
            sf[16 + r + 8 * laneHalf][nt * 16 + laneRow] = (__bf16)v1;
        }
}

// ---------------------------------------------------------------------------
// Main fused kernel: gather + 3 GEMMs + sigmoid gating. One wave = 32 rows.
// ---------------------------------------------------------------------------
__global__ __launch_bounds__(WPB_ * 32)
void fused_mlp_kernel(const float* __restrict__ feat1,
                      const float* __restrict__ feat2,
                      const int*   __restrict__ coords1,
                      const unsigned* __restrict__ T,
                      const __bf16* __restrict__ W1b,
                      const __bf16* __restrict__ W2b,
                      const __bf16* __restrict__ W3b,
                      float* __restrict__ out) {
    // per-wave private LDS tiles (row stride 264 bf16 = 528 B, 16B-aligned chunks)
    __shared__ __attribute__((aligned(16))) __bf16 sf1[WPB_][32][264];
    __shared__ __attribute__((aligned(16))) __bf16 sf2[WPB_][32][264];
    __shared__ int sidx[WPB_][32];

    const int wave = threadIdx.x >> 5;
    const int lane = threadIdx.x & 31;
    const int grp  = blockIdx.x * WPB_ + wave;
    if (grp >= GROUPS_) return;              // whole-wave exit; no block sync used

    const int laneRow  = lane & 15;
    const int laneHalf = lane >> 4;
    const int m0   = grp * 32;
    const bool full = (m0 + 32) <= N1_;      // wave-uniform tail flag

    // ---- resolve gather indices for this group's 32 rows ------------------
    {
        int m = m0 + lane; m = m < N1_ ? m : N1_ - 1;   // clamp tail (loads only)
        unsigned b = (unsigned)coords1[3*m + 0];
        unsigned y = (unsigned)coords1[3*m + 1];
        unsigned x = (unsigned)coords1[3*m + 2];
        unsigned key = b * (unsigned)HWl_ + y * (unsigned)WW_ + x;
        sidx[wave][lane] = (int)T[key];
    }
    lds_fence();

    int r0 = m0 + laneRow;
    int r1 = m0 + 16 + laneRow; r1 = r1 < N1_ ? r1 : N1_ - 1;
    const float* rowA1_0 = feat1 + (size_t)r0 * C_;
    const float* rowA1_1 = feat1 + (size_t)r1 * C_;
    const float* rowA2_0 = feat2 + (size_t)sidx[wave][laneRow] * C_;
    const float* rowA2_1 = feat2 + (size_t)sidx[wave][16 + laneRow] * C_;

    v8f acc0[16], acc1[16];
    const v8f vzero = {0.f,0.f,0.f,0.f,0.f,0.f,0.f,0.f};

    // ======================= GEMM1: f1 = relu(cat @ W1), K = 512 ==========
#pragma unroll
    for (int nt = 0; nt < 16; ++nt) { acc0[nt] = vzero; acc1[nt] = vzero; }

    // K 0..255 : mf1 rows (streamed once -> non-temporal)
    for (int s = 0; s < 8; ++s) {
        const int kbase = s * 32;
        v16bf a0 = load_a_f32<true>(rowA1_0 + kbase, laneHalf * 8);
        v16bf a1 = load_a_f32<true>(rowA1_1 + kbase, laneHalf * 8);
        v16bf bcur = load_b(W1b, kbase, 0, lane);
#pragma unroll
        for (int nt = 0; nt < 16; ++nt) {
            v16bf bnext = bcur;
            if (nt < 15) bnext = load_b(W1b, kbase, (nt + 1) * 16, lane);
            acc0[nt] = wmma_bf16(a0, bcur, acc0[nt]);
            acc1[nt] = wmma_bf16(a1, bcur, acc1[nt]);
            bcur = bnext;
        }
    }
    // K 256..511 : gathered mf2 rows (revisited in epilogue -> keep temporal)
    for (int s = 0; s < 8; ++s) {
        const int kbase = s * 32;
        v16bf a0 = load_a_f32<false>(rowA2_0 + kbase, laneHalf * 8);
        v16bf a1 = load_a_f32<false>(rowA2_1 + kbase, laneHalf * 8);
        v16bf bcur = load_b(W1b, 256 + kbase, 0, lane);
#pragma unroll
        for (int nt = 0; nt < 16; ++nt) {
            v16bf bnext = bcur;
            if (nt < 15) bnext = load_b(W1b, 256 + kbase, (nt + 1) * 16, lane);
            acc0[nt] = wmma_bf16(a0, bcur, acc0[nt]);
            acc1[nt] = wmma_bf16(a1, bcur, acc1[nt]);
            bcur = bnext;
        }
    }

    relu_store_lds(sf1[wave], acc0, acc1, laneRow, laneHalf);
    lds_fence();

    // ======================= GEMM2: f2 = relu(f1 @ W2), K = 256 ===========
#pragma unroll
    for (int nt = 0; nt < 16; ++nt) { acc0[nt] = vzero; acc1[nt] = vzero; }
    gemm_lds(sf1[wave], W2b, lane, laneRow, laneHalf, acc0, acc1);

    relu_store_lds(sf2[wave], acc0, acc1, laneRow, laneHalf);
    lds_fence();

    // ======================= GEMM3: f3 = f2 @ W3, K = 256 =================
#pragma unroll
    for (int nt = 0; nt < 16; ++nt) { acc0[nt] = vzero; acc1[nt] = vzero; }
    gemm_lds(sf2[wave], W3b, lane, laneRow, laneHalf, acc0, acc1);

    // ============ epilogue: att = sigmoid(f3); out = f1*att + mf2 ==========
#pragma unroll
    for (int t = 0; t < 2; ++t) {
        if (t == 1 && !full) break;          // wave-uniform tail skip
        const v8f* acc = (t == 0) ? acc0 : acc1;
#pragma unroll
        for (int nt = 0; nt < 16; ++nt) {
#pragma unroll
            for (int r = 0; r < 8; ++r) {
                const int m = t * 16 + r + 8 * laneHalf;
                const int n = nt * 16 + laneRow;
                float att = 1.f / (1.f + __expf(-acc[nt][r]));
                float f1v = (float)sf1[wave][m][n];
                float mf2 = feat2[(size_t)sidx[wave][m] * C_ + n];
                __builtin_nontemporal_store(f1v * att + mf2,
                                            out + (size_t)(m0 + m) * C_ + n);
            }
        }
    }
}

// ---------------------------------------------------------------------------
// Launch
// ---------------------------------------------------------------------------
extern "C" void kernel_launch(void* const* d_in, const int* in_sizes, int n_in,
                              void* d_out, int out_size, void* d_ws, size_t ws_size,
                              hipStream_t stream) {
    (void)in_sizes; (void)n_in; (void)out_size; (void)ws_size;

    const float* feat1   = (const float*)d_in[0];
    const float* feat2   = (const float*)d_in[1];
    const int*   coords1 = (const int*)d_in[2];
    const int*   coords2 = (const int*)d_in[3];
    const float* W1      = (const float*)d_in[4];
    const float* W2      = (const float*)d_in[5];
    const float* W3      = (const float*)d_in[6];
    float*       out     = (float*)d_out;

    unsigned* T   = (unsigned*)((char*)d_ws + WS_OFF_TABLE);
    __bf16*   W1b = (__bf16*)((char*)d_ws + WS_OFF_W1B);
    __bf16*   W2b = (__bf16*)((char*)d_ws + WS_OFF_W2B);
    __bf16*   W3b = (__bf16*)((char*)d_ws + WS_OFF_W3B);

    build_table_kernel<<<(N2_ + 255) / 256, 256, 0, stream>>>(coords2, T);
    cvt_weights_kernel<<<(512 * 256 + 255) / 256, 256, 0, stream>>>(W1, W2, W3,
                                                                    W1b, W2b, W3b);
    fused_mlp_kernel<<<(GROUPS_ + WPB_ - 1) / WPB_, WPB_ * 32, 0, stream>>>(
        feat1, feat2, coords1, T, W1b, W2b, W3b, out);
}
